// RandLANet_54082228191784
// MI455X (gfx1250) — matrix operand order repaired and checked by reading, not compiled, not run on hardware
//
#include <hip/hip_runtime.h>
#include <math.h>

typedef __attribute__((ext_vector_type(16))) __bf16 v16bf;
typedef __attribute__((ext_vector_type(8)))  float  v8f;

#define LRELU_F 0.2f
#define EPS_BN  1e-6f
#define BQ      2
#define KNN     16
#define LW      4   // waves per GEMM block

__device__ __forceinline__ unsigned short f2bf(float x) {
  unsigned int u = __float_as_uint(x);
  unsigned int r = u + 0x7FFFu + ((u >> 16) & 1u);   // round-to-nearest-even
  return (unsigned short)(r >> 16);
}

// ---------------------------------------------------------------------------
// Generic GEMM: Y[r,o] = sum_i X[r,i] * Wm[o,i] + bias[o]
// Block = 128 threads = 4 waves. Each wave owns one 16-row tile and NT
// 16-col tiles -> NT v_wmma per K-step off a single A fragment.
// LDS double-buffered: one barrier per K-step; stage(k+1) overlaps wmma(k).
// Branchless staging via clamped indices (loads always in-bounds).
// ---------------------------------------------------------------------------
template <int NT>
__global__ __launch_bounds__(128)
void k_linear_wmma(const float* __restrict__ X, const float* __restrict__ Wm,
                   const float* __restrict__ bias, float* __restrict__ Y,
                   int R, int Cin, int Cout)
{
  __shared__ unsigned short sA[2][LW * 16][32];
  __shared__ unsigned short sB[2][16 * NT][32];
  const int tid  = threadIdx.x;
  const int wave = tid >> 5;
  const int lane = tid & 31;
  const int rowBlk  = blockIdx.x * (LW * 16);
  const int rowBase = rowBlk + wave * 16;
  const int colBase = blockIdx.y * (16 * NT);
  const int m    = lane & 15;
  const int half = lane >> 4;

  v8f acc[NT];
#pragma unroll
  for (int nt = 0; nt < NT; ++nt) acc[nt] = {};

  auto stage = [&](int k0, int p) {
    // A tile: 64 rows x 32 k
    for (int t = tid; t < LW * 16 * 32; t += 128) {
      int rm = t >> 5;
      int kk = t & 31;
      int r = rowBlk + rm;
      int k = k0 + kk;
      int rc = r < R ? r : R - 1;
      int kc = k < Cin ? k : Cin - 1;
      float v = X[(long long)rc * Cin + kc];
      v = (r < R && k < Cin) ? v : 0.0f;
      sA[p][rm][kk] = f2bf(v);
    }
    // B tile: 16*NT cols x 32 k
    for (int t = tid; t < 16 * NT * 32; t += 128) {
      int n  = t >> 5;
      int kk = t & 31;
      int c = colBase + n;
      int k = k0 + kk;
      int cc = c < Cout ? c : Cout - 1;
      int kc = k < Cin ? k : Cin - 1;
      float v = Wm[(long long)cc * Cin + kc];
      v = (c < Cout && k < Cin) ? v : 0.0f;
      sB[p][n][kk] = f2bf(v);
    }
  };

  const int nk = (Cin + 31) >> 5;
  stage(0, 0);
  __syncthreads();
  for (int it = 0; it < nk; ++it) {
    const int p = it & 1;
    if (it + 1 < nk) stage((it + 1) * 32, p ^ 1);           // overlap next stage
    if (it + 2 < nk)                                        // L2 prefetch (global_prefetch_b8)
      __builtin_prefetch(X + (long long)(rowBlk + (tid >> 1)) * Cin + (it + 2) * 32 + (tid & 1) * 16, 0, 1);

    union { v16bf v; unsigned short u[16]; } fa, fb[NT];
#pragma unroll
    for (int vi = 0; vi < 8; ++vi) {
      int ka = ((vi >> 2) << 4) + (half << 3) + ((vi & 3) << 1);  // A K index
      fa.u[2 * vi]     = sA[p][wave * 16 + m][ka];
      fa.u[2 * vi + 1] = sA[p][wave * 16 + m][ka + 1];
      int kb = ((vi >> 2) << 4) + (half << 3) + ((vi & 3) << 1);  // B K index
#pragma unroll
      for (int nt = 0; nt < NT; ++nt) {
        fb[nt].u[2 * vi]     = sB[p][nt * 16 + m][kb];
        fb[nt].u[2 * vi + 1] = sB[p][nt * 16 + m][kb + 1];
      }
    }
#pragma unroll
    for (int nt = 0; nt < NT; ++nt)
      acc[nt] = __builtin_amdgcn_wmma_f32_16x16x32_bf16(false, fa.v, false, fb[nt].v,
                                                        (short)0, acc[nt], false, false);
    __syncthreads();
  }

#pragma unroll
  for (int nt = 0; nt < NT; ++nt) {
    int n = colBase + nt * 16 + m;
    if (n < Cout) {
      float bv = bias ? bias[n] : 0.0f;
#pragma unroll
      for (int r8 = 0; r8 < 8; ++r8) {
        int row = rowBase + r8 + (half << 3);   // C layout: M = vgpr + 8*half
        if (row < R) Y[(long long)row * Cout + n] = acc[nt][r8] + bv;
      }
    }
  }
}

// ---------------------------------------------------------------------------
// BatchNorm: per-channel mean / rsqrt(var+eps) over R rows (block per channel)
// ---------------------------------------------------------------------------
__global__ __launch_bounds__(256)
void k_bn_stats(const float* __restrict__ Y, long long R, int C,
                float* __restrict__ mean, float* __restrict__ istd)
{
  __shared__ float ss[256], sq[256];
  int c = blockIdx.x;
  float s = 0.f, q = 0.f;
  for (long long r = threadIdx.x; r < R; r += 256) {
    float v = Y[r * C + c];
    s += v; q += v * v;
  }
  ss[threadIdx.x] = s; sq[threadIdx.x] = q;
  __syncthreads();
  for (int o = 128; o > 0; o >>= 1) {
    if (threadIdx.x < o) {
      ss[threadIdx.x] += ss[threadIdx.x + o];
      sq[threadIdx.x] += sq[threadIdx.x + o];
    }
    __syncthreads();
  }
  if (threadIdx.x == 0) {
    float mu  = ss[0] / (float)R;
    float var = sq[0] / (float)R - mu * mu;
    mean[c] = mu;
    istd[c] = rsqrtf(fmaxf(var, 0.f) + EPS_BN);
  }
}

__global__ void k_bn_act(float* __restrict__ Y, long long total, int C,
                         const float* __restrict__ mean, const float* __restrict__ istd,
                         const float* __restrict__ g, const float* __restrict__ beta,
                         int act)
{
  long long t = blockIdx.x * (long long)blockDim.x + threadIdx.x;
  if (t >= total) return;
  int c = (int)(t % C);
  float v = (Y[t] - mean[c]) * istd[c] * g[c] + beta[c];
  if (act) v = v > 0.f ? v : LRELU_F * v;
  Y[t] = v;
}

// out[(b,n,k), coff+c] = F[b, idx[(b,n,k)], c]  (write into concat slot)
__global__ void k_gather_cat(const float* __restrict__ F, const int* __restrict__ idx,
                             float* __restrict__ Out, long long total,
                             int N, int K, int Nsrc, int C, int Cout, int coff)
{
  long long t = blockIdx.x * (long long)blockDim.x + threadIdx.x;
  if (t >= total) return;
  int c = (int)(t % C);
  long long r  = t / C;          // (b*N+n)*K+k
  long long bn = r / K;          // b*N+n
  int b = (int)(bn / N);
  int src = idx[r];
  Out[r * Cout + coff + c] = F[((long long)b * Nsrc + src) * C + c];
}

// Out[r, coff+c] = Src[r, c]
__global__ void k_copy_cat(const float* __restrict__ Src, float* __restrict__ Out,
                           long long total, int C, int Cout, int coff)
{
  long long t = blockIdx.x * (long long)blockDim.x + threadIdx.x;
  if (t >= total) return;
  int c = (int)(t % C);
  long long r = t / C;
  Out[r * Cout + coff + c] = Src[r * C + c];
}

// relpos: [dist, rel(3), tile(3), nb(3)] per (b,n,k)
__global__ void k_relpos(const float* __restrict__ xyz, const int* __restrict__ idx,
                         float* __restrict__ Out, long long total, int N, int K)
{
  long long t = blockIdx.x * (long long)blockDim.x + threadIdx.x;
  if (t >= total) return;
  long long bn = t / K;
  int b = (int)(bn / N);
  int n = (int)(bn % N);
  int src = idx[t];
  const float* pt = xyz + ((long long)b * N + n) * 3;
  const float* nb = xyz + ((long long)b * N + src) * 3;
  float rx = pt[0] - nb[0], ry = pt[1] - nb[1], rz = pt[2] - nb[2];
  float d = sqrtf(rx * rx + ry * ry + rz * rz);
  float* o = Out + t * 10;
  o[0] = d;  o[1] = rx;    o[2] = ry;    o[3] = rz;
  o[4] = pt[0]; o[5] = pt[1]; o[6] = pt[2];
  o[7] = nb[0]; o[8] = nb[1]; o[9] = nb[2];
}

// attention pool: softmax over K of logits, weighted sum of fset -> agg[b,n,d]
__global__ void k_attpool(const float* __restrict__ fset, const float* __restrict__ logits,
                          float* __restrict__ agg, long long total, int K, int D)
{
  long long t = blockIdx.x * (long long)blockDim.x + threadIdx.x;
  if (t >= total) return;
  int d = (int)(t % D);
  long long bn = t / D;
  const float* lg = logits + bn * (long long)K * D + d;
  const float* fs = fset   + bn * (long long)K * D + d;
  float mx = lg[0];
  for (int k = 1; k < KNN; ++k) mx = fmaxf(mx, lg[(long long)k * D]);
  float se = 0.f, acc = 0.f;
  for (int k = 0; k < KNN; ++k) {
    float e = __expf(lg[(long long)k * D] - mx);
    se  += e;
    acc += e * fs[(long long)k * D];
  }
  agg[t] = acc / se;
}

// Out[b,m,c] = max_k F[b, sub[b,m,k], c]
__global__ void k_gather_max(const float* __restrict__ F, const int* __restrict__ sub,
                             float* __restrict__ Out, long long total,
                             int M, int K, int Nsrc, int C)
{
  long long t = blockIdx.x * (long long)blockDim.x + threadIdx.x;
  if (t >= total) return;
  int c = (int)(t % C);
  long long bm = t / C;
  int b = (int)(bm / M);
  const int* id = sub + bm * K;
  const float* Fb = F + (long long)b * Nsrc * C + c;
  float mx = Fb[(long long)id[0] * C];
  for (int k = 1; k < KNN; ++k) mx = fmaxf(mx, Fb[(long long)id[k] * C]);
  Out[t] = mx;
}

// nearest interp into concat slot: Out[(b,n), coff+c] = F[b, interp[b,n], c]
__global__ void k_gather1_cat(const float* __restrict__ F, const int* __restrict__ interp,
                              float* __restrict__ Out, long long total,
                              int N, int Nsrc, int C, int Cout, int coff)
{
  long long t = blockIdx.x * (long long)blockDim.x + threadIdx.x;
  if (t >= total) return;
  int c = (int)(t % C);
  long long bn = t / C;
  int b = (int)(bn / N);
  int src = interp[bn];
  Out[bn * Cout + coff + c] = F[((long long)b * Nsrc + src) * C + c];
}

__global__ void k_add_lrelu(const float* __restrict__ A, const float* __restrict__ Bt,
                            float* __restrict__ Y, long long total)
{
  long long t = blockIdx.x * (long long)blockDim.x + threadIdx.x;
  if (t >= total) return;
  float v = A[t] + Bt[t];
  Y[t] = v > 0.f ? v : LRELU_F * v;
}

// ---------------------------------------------------------------------------
// Host orchestration
// ---------------------------------------------------------------------------
namespace {

struct Conv { const float *W, *b, *beta, *g; };

static inline unsigned gdiv(long long a, long long b) { return (unsigned)((a + b - 1) / b); }

struct Ctx { hipStream_t s; float* mean; float* istd; };

static void launch_linear(const Ctx& c, const float* X, const Conv& p, float* Y,
                          long long R, int Cin, int Cout)
{
  if (Cout >= 64) {
    dim3 grid(gdiv(R, 64), gdiv(Cout, 64));
    k_linear_wmma<4><<<grid, 128, 0, c.s>>>(X, p.W, p.b, Y, (int)R, Cin, Cout);
  } else if (Cout >= 32) {
    dim3 grid(gdiv(R, 64), gdiv(Cout, 32));
    k_linear_wmma<2><<<grid, 128, 0, c.s>>>(X, p.W, p.b, Y, (int)R, Cin, Cout);
  } else {
    dim3 grid(gdiv(R, 64), gdiv(Cout, 16));
    k_linear_wmma<1><<<grid, 128, 0, c.s>>>(X, p.W, p.b, Y, (int)R, Cin, Cout);
  }
}

static void launch_bn_act(const Ctx& c, float* Y, long long R, int C,
                          const float* g, const float* beta, bool act)
{
  k_bn_stats<<<dim3((unsigned)C), 256, 0, c.s>>>(Y, R, C, c.mean, c.istd);
  long long total = R * C;
  k_bn_act<<<gdiv(total, 256), 256, 0, c.s>>>(Y, total, C, c.mean, c.istd, g, beta, act ? 1 : 0);
}

static void launch_conv(const Ctx& c, const float* X, const Conv& p, float* Y,
                        long long R, int Cin, int Cout, bool act)
{
  launch_linear(c, X, p, Y, R, Cin, Cout);
  launch_bn_act(c, Y, R, Cout, p.g, p.beta, act);
}

} // namespace

extern "C" void kernel_launch(void* const* d_in, const int* in_sizes, int n_in,
                              void* d_out, int out_size, void* d_ws, size_t ws_size,
                              hipStream_t stream)
{
  (void)in_sizes; (void)out_size; (void)ws_size;
  if (n_in < 179) return;  // params pytree-leaf layout assumption not met

  static const int NS[5]   = {16384, 4096, 1024, 256, 64};
  static const int DOUT[4] = {16, 64, 128, 256};

  const float* xyz[4]; const int* neigh[4]; const int* sub[4]; const int* interp[4];
  for (int i = 0; i < 4; ++i) {
    xyz[i]    = (const float*)d_in[4 * i + 0];
    neigh[i]  = (const int*)  d_in[4 * i + 1];
    sub[i]    = (const int*)  d_in[4 * i + 2];
    interp[i] = (const int*)  d_in[4 * i + 3];
  }
  const float* features = (const float*)d_in[16];
  // params leaves, sorted-dict-key pytree order:
  const float* bn0_beta = (const float*)d_in[17];
  const float* bn0_g    = (const float*)d_in[18];
  auto convAt = [&](int idx) -> Conv {
    return Conv{(const float*)d_in[idx],     (const float*)d_in[idx + 1],
                (const float*)d_in[idx + 2], (const float*)d_in[idx + 3]};
  };
  Conv dec[4]; for (int j = 0; j < 4; ++j) dec[j] = convAt(19 + 4 * j);
  Conv dec0 = convAt(35);
  const int encBase = 39;                       // 4 blocks x 32 leaves -> 39..166
  Conv fc_final{(const float*)d_in[167], (const float*)d_in[168], nullptr, nullptr};
  Conv fc0{(const float*)d_in[169], (const float*)d_in[170], nullptr, nullptr};
  Conv fc1 = convAt(171);
  Conv fc2 = convAt(175);

  // ---- workspace bump allocator (floats) ----
  float* ws = (float*)d_ws;
  long long off = 0;
  auto alloc = [&](long long n) { float* p = ws + off; off += n; return p; };
  float* FE0 = alloc((long long)BQ * 16384 * 32);   // enc[0] skip
  float* FS[4];
  FS[0] = alloc((long long)BQ * 4096 * 32);
  FS[1] = alloc((long long)BQ * 1024 * 128);
  FS[2] = alloc((long long)BQ * 256 * 256);
  FS[3] = alloc((long long)BQ * 64 * 512);
  float* FA = alloc((long long)BQ * 16384 * 64);
  float* FB = alloc((long long)BQ * 16384 * 64);
  float* S1 = alloc((long long)BQ * 16384 * KNN * 10);  // relpos / fxyz2
  float* S2 = alloc((long long)BQ * 16384 * KNN * 8);   // fxyz (B*N*K*dh)
  float* S3 = alloc((long long)BQ * 16384 * KNN * 16);  // concat (B*N*K*2dh)
  float* S4 = alloc((long long)BQ * 16384 * KNN * 16);  // logits
  float* S5 = alloc((long long)BQ * 16384 * 16);        // agg (B*N*2dh)
  float* S6 = alloc((long long)BQ * 16384 * 8);         // att1 out (B*N*dh)
  float* T1 = alloc((long long)BQ * 16384 * 8);         // blk.mlp1 out
  float* T2 = alloc((long long)BQ * 16384 * 16);        // building-block out
  float* T3 = alloc((long long)BQ * 16384 * 32);        // mlp2 out
  float* T4 = alloc((long long)BQ * 16384 * 32);        // shortcut out
  float* MEAN = alloc(1024);
  float* ISTD = alloc(1024);
  Ctx c{stream, MEAN, ISTD};

  // ---- stem: fc0 + bn0 + lrelu ----
  long long R0 = (long long)BQ * NS[0];
  launch_linear(c, features, fc0, FA, R0, 3, 8);
  launch_bn_act(c, FA, R0, 8, bn0_g, bn0_beta, true);

  // ---- encoder ----
  const float* fcur = FA;
  int dfeat = 8;
  for (int i = 0; i < 4; ++i) {
    int N = NS[i];
    long long R  = (long long)BQ * N;
    long long RK = R * KNN;
    int dO = DOUT[i];
    int dh = dO / 2;
    int eb = encBase + 32 * i;
    Conv att1_fc {(const float*)d_in[eb + 0], (const float*)d_in[eb + 1], nullptr, nullptr};
    Conv att1_mlp = convAt(eb + 2);
    Conv att2_fc {(const float*)d_in[eb + 6], (const float*)d_in[eb + 7], nullptr, nullptr};
    Conv att2_mlp = convAt(eb + 8);
    Conv lfa_m1 = convAt(eb + 12);
    Conv lfa_m2 = convAt(eb + 16);
    Conv b_m1 = convAt(eb + 20);
    Conv b_m2 = convAt(eb + 24);
    Conv b_sc = convAt(eb + 28);

    launch_conv(c, fcur, b_m1, T1, R, dfeat, dh, true);                    // blk.mlp1
    k_relpos<<<gdiv(RK, 256), 256, 0, stream>>>(xyz[i], neigh[i], S1, RK, N, KNN);
    launch_conv(c, S1, lfa_m1, S2, RK, 10, dh, true);                      // f_xyz
    // cat1 = [gather(T1), f_xyz]
    k_gather_cat<<<gdiv(RK * dh, 256), 256, 0, stream>>>(T1, neigh[i], S3, RK * dh, N, KNN, N, dh, 2 * dh, 0);
    k_copy_cat<<<gdiv(RK * dh, 256), 256, 0, stream>>>(S2, S3, RK * dh, dh, 2 * dh, dh);
    launch_linear(c, S3, att1_fc, S4, RK, 2 * dh, 2 * dh);                 // logits
    k_attpool<<<gdiv(R * 2 * dh, 256), 256, 0, stream>>>(S3, S4, S5, R * 2 * dh, KNN, 2 * dh);
    launch_conv(c, S5, att1_mlp, S6, R, 2 * dh, dh, true);                 // f
    launch_conv(c, S2, lfa_m2, S1, RK, dh, dh, true);                      // f_xyz2
    // cat2 = [gather(f), f_xyz2]
    k_gather_cat<<<gdiv(RK * dh, 256), 256, 0, stream>>>(S6, neigh[i], S3, RK * dh, N, KNN, N, dh, 2 * dh, 0);
    k_copy_cat<<<gdiv(RK * dh, 256), 256, 0, stream>>>(S1, S3, RK * dh, dh, 2 * dh, dh);
    launch_linear(c, S3, att2_fc, S4, RK, 2 * dh, 2 * dh);
    k_attpool<<<gdiv(R * 2 * dh, 256), 256, 0, stream>>>(S3, S4, S5, R * 2 * dh, KNN, 2 * dh);
    launch_conv(c, S5, att2_mlp, T2, R, 2 * dh, dO, true);                 // building block out
    launch_conv(c, T2, b_m2, T3, R, dO, 2 * dO, false);                    // mlp2 (no act)
    launch_conv(c, fcur, b_sc, T4, R, dfeat, 2 * dO, false);               // shortcut (no act)
    float* fe = (i == 0) ? FE0 : FB;
    k_add_lrelu<<<gdiv(R * 2 * dO, 256), 256, 0, stream>>>(T3, T4, fe, R * (long long)(2 * dO));
    int M = NS[i + 1];
    long long tot = (long long)BQ * M * 2 * dO;
    k_gather_max<<<gdiv(tot, 256), 256, 0, stream>>>(fe, sub[i], FS[i], tot, M, KNN, N, 2 * dO);
    fcur = FS[i];
    dfeat = 2 * dO;
  }

  // ---- decoder ----
  launch_conv(c, FS[3], dec0, FA, (long long)BQ * NS[4], 512, 512, true);
  fcur = FA;
  int fch = 512;
  const float* skips[4] = {FS[2], FS[1], FS[0], FE0};
  const int skipC[4]    = {256, 128, 32, 32};
  for (int j = 0; j < 4; ++j) {
    int lvl = 3 - j;
    int N = NS[lvl], Nsrc = NS[lvl + 1];
    long long R = (long long)BQ * N;
    int cin = skipC[j] + fch, cout = skipC[j];
    k_copy_cat<<<gdiv(R * skipC[j], 256), 256, 0, stream>>>(skips[j], S3, R * skipC[j], skipC[j], cin, 0);
    k_gather1_cat<<<gdiv(R * fch, 256), 256, 0, stream>>>(fcur, interp[lvl], S3, R * fch, N, Nsrc, fch, cin, skipC[j]);
    float* dst = (j % 2 == 0) ? FB : FA;
    launch_conv(c, S3, dec[j], dst, R, cin, cout, true);
    fcur = dst;
    fch = cout;
  }

  // ---- head ----
  long long Rh = (long long)BQ * NS[0];
  launch_conv(c, fcur, fc1, FB, Rh, 32, 64, true);
  launch_conv(c, FB, fc2, FA, Rh, 64, 32, true);
  launch_linear(c, FA, fc_final, (float*)d_out, Rh, 32, 13);
}
